// BGRL_28269474742837
// MI455X (gfx1250) — matrix operand order repaired and checked by reading, not compiled, not run on hardware
//
#include <hip/hip_runtime.h>

#define D_IN   256
#define D_HID  128
#define D_OUT_ 128
#define P_HID  512

typedef __bf16 bf16_t;
typedef __attribute__((ext_vector_type(16))) __bf16 v16bf;
typedef __attribute__((ext_vector_type(8)))  float  v8f;

// ---------------------------------------------------------------------------
// Weight prep: W[K x Nout] (f32, row-major) -> WT[Nout x K] (bf16, row-major)
// ---------------------------------------------------------------------------
__global__ __launch_bounds__(256) void k_wtrans(const float* __restrict__ W,
                                                bf16_t* __restrict__ WT,
                                                int K, int Nout) {
  int i = blockIdx.x * 256 + threadIdx.x;
  if (i < K * Nout) {
    int n = i / K, k = i - n * K;
    WT[i] = (bf16_t)W[(size_t)k * Nout + n];
  }
}

// ---------------------------------------------------------------------------
// WMMA GEMM: OUT[M x NoutTotal] = A[M x K] (f32) * W (via WT bf16 Nout x K)
// block = 256 thr = 8 waves; block tile 128(M) x 128(N); wave tile 16 x 128.
// W slice staged to LDS via async global->LDS (ASYNCcnt); B fragments
// preloaded in bulk so WMMAs retire against partial s_wait_dscnt.
// Optional fused epilogue: +bias[col], PReLU(aprelu).
// ---------------------------------------------------------------------------
template<int K>
__global__ __launch_bounds__(256) void k_gemm_wmma(
    const float* __restrict__ A, const bf16_t* __restrict__ WT,
    float* __restrict__ OUT, int M, int NoutTotal,
    const float* __restrict__ bias, const float* __restrict__ aprelu)
{
  __shared__ bf16_t lds[128 * 128];          // 32 KB: 128 W-cols x 128 K-chunk
  const int tid  = threadIdx.x;
  const int wave = tid >> 5;
  const int lane = tid & 31;
  const int rowBase = blockIdx.x * 128 + wave * 16;
  const int colBase = blockIdx.y * 128;

  v8f acc[8];
  #pragma unroll
  for (int j = 0; j < 8; ++j)
    #pragma unroll
    for (int v = 0; v < 8; ++v) acc[j][v] = 0.0f;

  int mClamp = rowBase + (lane & 15);
  if (mClamp > M - 1) mClamp = M - 1;
  const float* Arow = A + (size_t)mClamp * K;
  const int koffA = (lane >= 16) ? 16 : 0;   // A frag: lanes0-15 K 0..15,      lanes16-31 K 16..31
  const int koffB = (lane >= 16) ? 16 : 0;   // B frag: lanes0-15 K 0..15,      lanes16-31 K 16..31

  for (int kc = 0; kc < K; kc += 128) {
    // async cooperative load: WT slice [colBase..+128) x [kc..+128) -> LDS
    #pragma unroll
    for (int i = 0; i < 8; ++i) {            // 2048 x 16B / 256 threads
      int idx = tid + i * 256;
      int row = idx >> 4;
      int off = idx & 15;
      unsigned long long gaddr = (unsigned long long)(size_t)
          ((const char*)(WT + (size_t)(colBase + row) * K + kc) + off * 16);
      unsigned ldsoff = (unsigned)(size_t)((char*)lds + idx * 16);
      asm volatile("global_load_async_to_lds_b128 %0, %1, off"
                   :: "v"(ldsoff), "v"(gaddr) : "memory");
    }
    asm volatile("s_wait_asynccnt 0x0" ::: "memory");
    __syncthreads();
    if (kc + 128 < K) __builtin_prefetch(Arow + kc + 128, 0, 0);

    #pragma unroll
    for (int ks = 0; ks < 128; ks += 32) {
      const float* ap = Arow + kc + ks + koffA;
      float4 q0 = *(const float4*)(ap);
      float4 q1 = *(const float4*)(ap + 4);
      float4 q2 = *(const float4*)(ap + 8);
      float4 q3 = *(const float4*)(ap + 12);
      v16bf afrag;
      afrag[0]=(bf16_t)q0.x;  afrag[1]=(bf16_t)q0.y;  afrag[2]=(bf16_t)q0.z;  afrag[3]=(bf16_t)q0.w;
      afrag[4]=(bf16_t)q1.x;  afrag[5]=(bf16_t)q1.y;  afrag[6]=(bf16_t)q1.z;  afrag[7]=(bf16_t)q1.w;
      afrag[8]=(bf16_t)q2.x;  afrag[9]=(bf16_t)q2.y;  afrag[10]=(bf16_t)q2.z; afrag[11]=(bf16_t)q2.w;
      afrag[12]=(bf16_t)q3.x; afrag[13]=(bf16_t)q3.y; afrag[14]=(bf16_t)q3.z; afrag[15]=(bf16_t)q3.w;

      v16bf bfr[8];
      #pragma unroll
      for (int j = 0; j < 8; ++j)
        bfr[j] = *(const v16bf*)(lds + (j * 16 + (lane & 15)) * 128 + ks + koffB);
      #pragma unroll
      for (int j = 0; j < 8; ++j)
        acc[j] = __builtin_amdgcn_wmma_f32_16x16x32_bf16(
            false, afrag, false, bfr[j], (short)0, acc[j], false, false);
    }
    __syncthreads();
  }

  const float aval = aprelu ? aprelu[0] : 0.0f;
  const int mOff = (lane >= 16) ? 8 : 0;     // C layout: VGPR v -> M = v (+8 for upper lanes)
  #pragma unroll
  for (int j = 0; j < 8; ++j) {
    const int n = colBase + j * 16 + (lane & 15);
    const float bval = bias ? bias[n] : 0.0f;
    #pragma unroll
    for (int v = 0; v < 8; ++v) {
      int m = rowBase + v + mOff;
      if (m < M) {
        float y = acc[j][v] + bval;
        if (aprelu) y = (y >= 0.0f) ? y : aval * y;
        OUT[(size_t)m * NoutTotal + n] = y;
      }
    }
  }
}

// ---------------------------------------------------------------------------
// Degree / normalization
// ---------------------------------------------------------------------------
__global__ __launch_bounds__(256) void k_setone(float* p, int n) {
  int i = blockIdx.x * 256 + threadIdx.x;
  if (i < n) p[i] = 1.0f;                    // self-loop contributes 1 to deg
}
__global__ __launch_bounds__(256) void k_degcount(const int* __restrict__ dst,
                                                  float* __restrict__ deg, int E) {
  int e = blockIdx.x * 256 + threadIdx.x;
  if (e < E) atomicAdd(&deg[dst[e]], 1.0f);
}
__global__ __launch_bounds__(256) void k_rsqrt(float* p, int n) {
  int i = blockIdx.x * 256 + threadIdx.x;
  if (i < n) p[i] = rsqrtf(p[i]);            // deg >= 1 always
}
__global__ __launch_bounds__(256) void k_zero(float* p, size_t n) {
  size_t i = (size_t)blockIdx.x * 256 + threadIdx.x;
  if (i < n) p[i] = 0.0f;
}

// ---------------------------------------------------------------------------
// GCN propagation: S = self-loop init, then atomic edge scatter
// ---------------------------------------------------------------------------
__global__ __launch_bounds__(256) void k_selfloop(const float* __restrict__ H,
                                                  const float* __restrict__ dinv,
                                                  float* __restrict__ S, int N) {
  size_t i = (size_t)blockIdx.x * 256 + threadIdx.x;
  if (i < (size_t)N * D_HID) {
    int node = (int)(i >> 7);                // D = 128
    float w = dinv[node];
    S[i] = H[i] * w * w;
  }
}
__global__ __launch_bounds__(256) void k_scatter(const int* __restrict__ src,
                                                 const int* __restrict__ dst,
                                                 const float* __restrict__ dinv,
                                                 const float* __restrict__ H,
                                                 float* __restrict__ S, int E) {
  int gid  = blockIdx.x * 256 + threadIdx.x;
  int e    = gid >> 5;
  int lane = gid & 31;
  if (e < E) {
    int s = src[e], d = dst[e];
    float w = dinv[s] * dinv[d];
    const float* hs = H + (size_t)s * D_HID;
    float*       od = S + (size_t)d * D_HID;
    #pragma unroll
    for (int f = 0; f < D_HID; f += 32)
      atomicAdd(&od[f + lane], hs[f + lane] * w);
  }
}

// ---------------------------------------------------------------------------
// BatchNorm (training-mode batch stats, D = 128) + PReLU
// ---------------------------------------------------------------------------
__global__ __launch_bounds__(256) void k_colstats(const float* __restrict__ X, int Nrows,
                                                  float* __restrict__ sum,
                                                  float* __restrict__ sumsq) {
  __shared__ float ls[256], ls2[256];
  int col    = threadIdx.x & 127;
  int rowOff = threadIdx.x >> 7;             // 2 rows per block-iteration
  float s = 0.0f, s2 = 0.0f;
  for (int r = blockIdx.x * 2 + rowOff; r < Nrows; r += gridDim.x * 2) {
    float v = X[(size_t)r * 128 + col];
    s += v; s2 += v * v;
  }
  ls[threadIdx.x] = s; ls2[threadIdx.x] = s2;
  __syncthreads();
  if (rowOff == 0) {
    s += ls[col + 128]; s2 += ls2[col + 128];
    atomicAdd(&sum[col], s);
    atomicAdd(&sumsq[col], s2);
  }
}
__global__ void k_bnfin(const float* sum, const float* sumsq,
                        float* mu, float* rstd, int Nrows) {
  int c = threadIdx.x;
  if (c < 128) {
    float m = sum[c] / (float)Nrows;
    mu[c] = m;
    float v = sumsq[c] / (float)Nrows - m * m;
    rstd[c] = rsqrtf(v + 1e-5f);
  }
}
__global__ __launch_bounds__(256) void k_bnapply(const float* __restrict__ X,
                                                 float* __restrict__ Y,
                                                 const float* __restrict__ mu,
                                                 const float* __restrict__ rstd,
                                                 const float* __restrict__ g,
                                                 const float* __restrict__ be,
                                                 const float* __restrict__ a, int N) {
  size_t i = (size_t)blockIdx.x * 256 + threadIdx.x;
  if (i < (size_t)N * 128) {
    int c = (int)(i & 127);
    float y = (X[i] - mu[c]) * rstd[c] * g[c] + be[c];
    float av = a[0];
    Y[i] = (y >= 0.0f) ? y : av * y;
  }
}

// ---------------------------------------------------------------------------
// Cosine loss: one wave per node, accumulate mean(2 - 2*cos) into scalar
// ---------------------------------------------------------------------------
__global__ __launch_bounds__(256) void k_loss(const float* __restrict__ P,
                                              const float* __restrict__ T,
                                              float* __restrict__ lossOut, int N) {
  int gid  = blockIdx.x * 256 + threadIdx.x;
  int node = gid >> 5;
  int lane = gid & 31;
  if (node >= N) return;
  const float* p = P + (size_t)node * D_OUT_;
  const float* t = T + (size_t)node * D_OUT_;
  float dot = 0.0f, na = 0.0f, nb = 0.0f;
  #pragma unroll
  for (int f = 0; f < D_OUT_; f += 32) {
    float x = p[f + lane], y = t[f + lane];
    dot += x * y; na += x * x; nb += y * y;
  }
  #pragma unroll
  for (int m = 16; m >= 1; m >>= 1) {
    dot += __shfl_xor(dot, m, 32);
    na  += __shfl_xor(na,  m, 32);
    nb  += __shfl_xor(nb,  m, 32);
  }
  if (lane == 0) {
    float denom = fmaxf(sqrtf(na), 1e-12f) * fmaxf(sqrtf(nb), 1e-12f);
    float l = 2.0f - 2.0f * dot / denom;
    atomicAdd(lossOut, l / (float)N);
  }
}

// ---------------------------------------------------------------------------
// Host orchestration
// ---------------------------------------------------------------------------
extern "C" void kernel_launch(void* const* d_in, const int* in_sizes, int n_in,
                              void* d_out, int out_size, void* d_ws, size_t ws_size,
                              hipStream_t stream) {
  (void)n_in; (void)out_size; (void)ws_size;
  const float* x1  = (const float*)d_in[0];
  const float* x2  = (const float*)d_in[1];
  const int*   ei1 = (const int*)d_in[2];
  const int*   ei2 = (const int*)d_in[3];
  const int N = in_sizes[0] / D_IN;
  const int E = in_sizes[2] / 2;

  const float* sW1 = (const float*)d_in[4];
  const float* sg1 = (const float*)d_in[6];  const float* sbe1 = (const float*)d_in[7];
  const float* sa1 = (const float*)d_in[8];
  const float* sW2 = (const float*)d_in[9];
  const float* sg2 = (const float*)d_in[11]; const float* sbe2 = (const float*)d_in[12];
  const float* sa2 = (const float*)d_in[13];
  const float* tW1 = (const float*)d_in[14];
  const float* tg1 = (const float*)d_in[16]; const float* tbe1 = (const float*)d_in[17];
  const float* ta1 = (const float*)d_in[18];
  const float* tW2 = (const float*)d_in[19];
  const float* tg2 = (const float*)d_in[21]; const float* tbe2 = (const float*)d_in[22];
  const float* ta2 = (const float*)d_in[23];
  const float* pW1 = (const float*)d_in[24]; const float* pb1 = (const float*)d_in[25];
  const float* pa  = (const float*)d_in[26];
  const float* pW2 = (const float*)d_in[27]; const float* pb2 = (const float*)d_in[28];

  // workspace carve-out
  char* ws = (char*)d_ws;
  size_t off = 0;
  auto alloc = [&](size_t bytes) -> char* {
    off = (off + 255) & ~(size_t)255;
    char* p = ws + off; off += bytes; return p;
  };
  bf16_t* wt_sW1 = (bf16_t*)alloc((size_t)D_IN  * D_HID  * 2);
  bf16_t* wt_sW2 = (bf16_t*)alloc((size_t)D_HID * D_OUT_ * 2);
  bf16_t* wt_tW1 = (bf16_t*)alloc((size_t)D_IN  * D_HID  * 2);
  bf16_t* wt_tW2 = (bf16_t*)alloc((size_t)D_HID * D_OUT_ * 2);
  bf16_t* wt_pW1 = (bf16_t*)alloc((size_t)D_OUT_ * P_HID  * 2);
  bf16_t* wt_pW2 = (bf16_t*)alloc((size_t)P_HID  * D_OUT_ * 2);
  float* dinv1 = (float*)alloc((size_t)N * 4);
  float* dinv2 = (float*)alloc((size_t)N * 4);
  float* sumb  = (float*)alloc(128 * 4);
  float* sumsq = (float*)alloc(128 * 4);
  float* mu    = (float*)alloc(128 * 4);
  float* rstd  = (float*)alloc(128 * 4);
  float* Hbuf  = (float*)alloc((size_t)N * P_HID * 4);   // GEMM out / predictor hidden
  float* Sbuf  = (float*)alloc((size_t)N * D_HID * 4);   // scatter accumulator
  float* Y1buf = (float*)alloc((size_t)N * D_HID * 4);   // layer-1 activations
  float* Tbuf  = (float*)alloc((size_t)N * D_OUT_ * 4);  // teacher output (reused)
  float* Pbuf  = (float*)alloc((size_t)N * D_OUT_ * 4);  // predictor output (reused)

  float* out_v1 = (float*)d_out;
  float* out_v2 = out_v1 + (size_t)N * D_OUT_;
  float* d_loss = out_v1 + (size_t)2 * N * D_OUT_;

  const int gN    = (N + 255) / 256;
  const int gE    = (E + 255) / 256;
  const int gND   = (int)(((size_t)N * 128 + 255) / 256);
  const int gEw   = (int)(((size_t)E * 32 + 255) / 256);
  const int gNw   = (int)(((size_t)N * 32 + 255) / 256);
  const dim3 blk(256);

  // 1. weight prep (bf16 transpose)
  auto wtrans = [&](const float* W, bf16_t* WT, int K, int Nout) {
    k_wtrans<<<(K * Nout + 255) / 256, blk, 0, stream>>>(W, WT, K, Nout);
  };
  wtrans(sW1, wt_sW1, D_IN,  D_HID);
  wtrans(sW2, wt_sW2, D_HID, D_OUT_);
  wtrans(tW1, wt_tW1, D_IN,  D_HID);
  wtrans(tW2, wt_tW2, D_HID, D_OUT_);
  wtrans(pW1, wt_pW1, D_OUT_, P_HID);
  wtrans(pW2, wt_pW2, P_HID,  D_OUT_);

  // 2. symmetric GCN normalization per view
  k_setone<<<gN, blk, 0, stream>>>(dinv1, N);
  k_degcount<<<gE, blk, 0, stream>>>(ei1 + E, dinv1, E);
  k_rsqrt<<<gN, blk, 0, stream>>>(dinv1, N);
  k_setone<<<gN, blk, 0, stream>>>(dinv2, N);
  k_degcount<<<gE, blk, 0, stream>>>(ei2 + E, dinv2, E);
  k_rsqrt<<<gN, blk, 0, stream>>>(dinv2, N);

  k_zero<<<1, blk, 0, stream>>>(d_loss, 1);

  auto gemm = [&](int K, const float* A, const bf16_t* WT, float* OUT, int Nout,
                  const float* bias, const float* ap) {
    dim3 grid((N + 127) / 128, Nout / 128);
    if (K == 256)      k_gemm_wmma<256><<<grid, blk, 0, stream>>>(A, WT, OUT, N, Nout, bias, ap);
    else if (K == 128) k_gemm_wmma<128><<<grid, blk, 0, stream>>>(A, WT, OUT, N, Nout, bias, ap);
    else               k_gemm_wmma<512><<<grid, blk, 0, stream>>>(A, WT, OUT, N, Nout, bias, ap);
  };

  auto gcn_post = [&](const int* src, const int* dst, const float* dinv,
                      const float* g, const float* be, const float* a, float* Y) {
    k_selfloop<<<gND, blk, 0, stream>>>(Hbuf, dinv, Sbuf, N);
    k_scatter<<<gEw, blk, 0, stream>>>(src, dst, dinv, Hbuf, Sbuf, E);
    k_zero<<<1, blk, 0, stream>>>(sumb, 128);
    k_zero<<<1, blk, 0, stream>>>(sumsq, 128);
    k_colstats<<<1024, blk, 0, stream>>>(Sbuf, N, sumb, sumsq);
    k_bnfin<<<1, 128, 0, stream>>>(sumb, sumsq, mu, rstd, N);
    k_bnapply<<<gND, blk, 0, stream>>>(Sbuf, Y, mu, rstd, g, be, a, N);
  };

  auto encoder = [&](const float* X, const int* src, const int* dst, const float* dinv,
                     const bf16_t* WT1, const bf16_t* WT2,
                     const float* g1, const float* be1, const float* a1,
                     const float* g2, const float* be2, const float* a2, float* outp) {
    gemm(256, X, WT1, Hbuf, 128, nullptr, nullptr);      // bias dropped: BN removes it
    gcn_post(src, dst, dinv, g1, be1, a1, Y1buf);
    gemm(128, Y1buf, WT2, Hbuf, 128, nullptr, nullptr);
    gcn_post(src, dst, dinv, g2, be2, a2, outp);
  };

  auto predictor = [&](const float* V, float* Pout) {
    gemm(128, V, wt_pW1, Hbuf, 512, pb1, pa);            // fused bias + PReLU
    gemm(512, Hbuf, wt_pW2, Pout, 128, pb2, nullptr);    // fused bias
  };

  const int* s1 = ei1;       const int* d1 = ei1 + E;
  const int* s2 = ei2;       const int* d2 = ei2 + E;

  // student encoders -> outputs
  encoder(x1, s1, d1, dinv1, wt_sW1, wt_sW2, sg1, sbe1, sa1, sg2, sbe2, sa2, out_v1);
  encoder(x2, s2, d2, dinv2, wt_sW1, wt_sW2, sg1, sbe1, sa1, sg2, sbe2, sa2, out_v2);

  // loss1 = cos-loss(pred(v1_s), teacher(x2))
  encoder(x2, s2, d2, dinv2, wt_tW1, wt_tW2, tg1, tbe1, ta1, tg2, tbe2, ta2, Tbuf);
  predictor(out_v1, Pbuf);
  k_loss<<<gNw, blk, 0, stream>>>(Pbuf, Tbuf, d_loss, N);

  // loss2 = cos-loss(pred(v2_s), teacher(x1))   (reuse Tbuf/Pbuf)
  encoder(x1, s1, d1, dinv1, wt_tW1, wt_tW2, tg1, tbe1, ta1, tg2, tbe2, ta2, Tbuf);
  predictor(out_v2, Pbuf);
  k_loss<<<gNw, blk, 0, stream>>>(Pbuf, Tbuf, d_loss, N);
}
